// MediatorAttention_962072674724
// MI455X (gfx1250) — matrix operand order repaired and checked by reading, not compiled, not run on hardware
//
#include <hip/hip_runtime.h>
#include <stdint.h>

typedef __bf16 bf16;
typedef __attribute__((ext_vector_type(16))) __bf16 v16bf;
typedef __attribute__((ext_vector_type(8)))  float  v8f;
typedef __attribute__((ext_vector_type(4)))  int    v4i;

#define DEV __device__ __forceinline__

// ---------------- async global->LDS (CDNA5, ASYNCcnt) ----------------
// Builtin signature (from toolchain diagnostics): param1 = v4i* in AS1 (global),
// param2 = LDS side, then imm offset + imm cpol. Flat shared pointers carry the
// LDS offset in their low 32 bits (ISA 10.2), so build AS pointers via integer
// casts (avoids illegal generic->AS pointer casts).
#if defined(__has_builtin)
#if __has_builtin(__builtin_amdgcn_global_load_async_to_lds_b128)
#define HAVE_ASYNC_LDS 1
#endif
#endif

#ifdef HAVE_ASYNC_LDS
#define LDSP(p) ((__attribute__((address_space(3))) v4i*)(uint32_t)(uintptr_t)(p))
#define GLBP(p) ((__attribute__((address_space(1))) v4i*)(uintptr_t)(p))
#define ASYNC_COPY_B128(lds, gl) \
  __builtin_amdgcn_global_load_async_to_lds_b128(GLBP(gl), LDSP(lds), 0, 0)
#if __has_builtin(__builtin_amdgcn_s_wait_asynccnt)
#define ASYNC_WAIT() __builtin_amdgcn_s_wait_asynccnt(0)
#else
#define ASYNC_WAIT() asm volatile("s_wait_asynccnt 0" ::: "memory")
#endif
#else
#define ASYNC_COPY_B128(lds, gl) (*(uint4*)(lds) = *(const uint4*)(gl))
#define ASYNC_WAIT() do {} while (0)
#endif

// ---------------- helpers ----------------
DEV bf16 f2bf(float f) {
  union { float f; uint32_t u; } a; a.f = f;
  uint32_t r = a.u + 0x7FFFu + ((a.u >> 16) & 1u);
  union { uint16_t u; bf16 b; } o; o.u = (uint16_t)(r >> 16);
  return o.b;
}

// A fragment: 16x32 bf16 tile, row-major (stride in elems). k = 16*(lane/16) + e.
DEV v16bf load_a_frag(const bf16* base, int stride) {
  const int lane = threadIdx.x & 31;
  const int m = lane & 15, kg = lane >> 4;
  v16bf f;
#pragma unroll
  for (int p = 0; p < 8; ++p) {
    const int k = kg * 16 + p * 2;
    f[2 * p]     = base[m * stride + k];
    f[2 * p + 1] = base[m * stride + k + 1];
  }
  return f;
}

// B fragment: 32x16 tile, source row-major [k][n].
DEV v16bf load_b_frag(const bf16* base, int stride) {
  const int lane = threadIdx.x & 31;
  const int n = lane & 15, kg = lane >> 4;
  v16bf f;
#pragma unroll
  for (int e = 0; e < 16; ++e)
    f[e] = base[(kg * 16 + e) * stride + n];
  return f;
}

// B fragment from transposed source: row-major [n][k] 16x32 tile, used as B[k][n].
DEV v16bf load_bT_frag(const bf16* base, int stride) {
  const int lane = threadIdx.x & 31;
  const int n = lane & 15, kg = lane >> 4;
  v16bf f;
#pragma unroll
  for (int e = 0; e < 16; ++e)
    f[e] = base[n * stride + kg * 16 + e];
  return f;
}

DEV v8f wmma_bf16(v16bf a, v16bf b, v8f c) {
  return __builtin_amdgcn_wmma_f32_16x16x32_bf16(
      false, a, false, b, (short)0, c, false, false);
}

// ---------------- fp32 -> bf16 ----------------
__global__ void k_f32_to_bf16(const float* __restrict__ src,
                              bf16* __restrict__ dst, int n) {
  for (int i = blockIdx.x * blockDim.x + threadIdx.x; i < n;
       i += gridDim.x * blockDim.x)
    dst[i] = f2bf(src[i]);
}

// ---------------- WMMA GEMM: C[Mr,N] = A[Mr,K] @ W[K,N] (+bias) ----------------
// 256 threads = 8 waves. Block tile 128x128, BK=32, double-buffered async LDS.
// Wave tile 64x32 -> 4x2 accumulators.
template <bool OUT_BF16, bool BIAS>
__global__ __launch_bounds__(256) void k_gemm(const bf16* __restrict__ A,
                                              const bf16* __restrict__ W,
                                              const float* __restrict__ bias,
                                              void* __restrict__ Cout,
                                              int Mr, int N, int K) {
  __shared__ __attribute__((aligned(16))) bf16 As[2][128][40];  // 32+8 pad
  __shared__ __attribute__((aligned(16))) bf16 Bs[2][32][136];  // 128+8 pad
  const int tid = threadIdx.x;
  const int lane = tid & 31;
  const int w = tid >> 5;
  const int m0 = blockIdx.y * 128, n0 = blockIdx.x * 128;
  const int wr = (w >> 2) * 64, wc = (w & 3) * 32;

  v8f acc[4][2] = {};

  auto load_tiles = [&](int buf, int kk0) {
#pragma unroll
    for (int t = 0; t < 2; ++t) {  // A tile 128x32 = 512 x b128
      const int idx = tid + t * 256;
      const int r = idx >> 2, c = (idx & 3) * 8;
      ASYNC_COPY_B128(&As[buf][r][c], &A[(size_t)(m0 + r) * K + kk0 + c]);
    }
#pragma unroll
    for (int t = 0; t < 2; ++t) {  // B tile 32x128 = 512 x b128
      const int idx = tid + t * 256;
      const int r = idx >> 4, c = (idx & 15) * 8;
      ASYNC_COPY_B128(&Bs[buf][r][c], &W[(size_t)(kk0 + r) * N + n0 + c]);
    }
  };

  load_tiles(0, 0);
  ASYNC_WAIT();
  __syncthreads();

  int buf = 0;
  for (int kk0 = 0; kk0 < K; kk0 += 32) {
    if (kk0 + 32 < K) load_tiles(buf ^ 1, kk0 + 32);  // prefetch next tile
    v16bf af[4], bfm[2];
#pragma unroll
    for (int mi = 0; mi < 4; ++mi)
      af[mi] = load_a_frag(&As[buf][wr + mi * 16][0], 40);
    bfm[0] = load_b_frag(&Bs[buf][0][wc], 136);
    bfm[1] = load_b_frag(&Bs[buf][0][wc + 16], 136);
#pragma unroll
    for (int mi = 0; mi < 4; ++mi)
#pragma unroll
      for (int ni = 0; ni < 2; ++ni)
        acc[mi][ni] = wmma_bf16(af[mi], bfm[ni], acc[mi][ni]);
    ASYNC_WAIT();
    __syncthreads();
    buf ^= 1;
  }

  const int rbase = (lane >> 4) * 8, cin = lane & 15;
#pragma unroll
  for (int mi = 0; mi < 4; ++mi)
#pragma unroll
    for (int ni = 0; ni < 2; ++ni)
#pragma unroll
      for (int r = 0; r < 8; ++r) {
        const int row = m0 + wr + mi * 16 + rbase + r;
        const int col = n0 + wc + ni * 16 + cin;
        float v = acc[mi][ni][r];
        if (BIAS) v += bias[col];
        if (OUT_BF16)
          ((bf16*)Cout)[(size_t)row * N + col] = f2bf(v);
        else
          ((float*)Cout)[(size_t)row * N + col] = v;
      }
}

// ---------------- stage 1: mk_v[b,h] = softmax(med k^T / sqrt(d)) @ v ----------------
__global__ __launch_bounds__(128) void k_stage1(const bf16* __restrict__ medbf, // [64,1024]
                                                const bf16* __restrict__ kvbf,  // [B*L,2048]
                                                bf16* __restrict__ mkvbf) {     // [64*64,64]
  const int bh = blockIdx.x, b = bh >> 4, h = bh & 15;
  const float scale = 0.125f;
  __shared__ __attribute__((aligned(16))) bf16 medS[64][72];
  __shared__ __attribute__((aligned(16))) bf16 kS[64][72];
  __shared__ __attribute__((aligned(16))) bf16 vS[64][72];
  __shared__ __attribute__((aligned(16))) bf16 Ps[64][72];
  __shared__ __attribute__((aligned(16))) float Sg[64][72];
  __shared__ float rmax[64], rsum[64], rinv[64];
  const int tid = threadIdx.x, lane = tid & 31, w = tid >> 5;
  const int qr = (w >> 1) * 32, qc = (w & 1) * 32;
  const int rbase = (lane >> 4) * 8, cin = lane & 15;

#pragma unroll
  for (int t = 0; t < 4; ++t) {
    const int idx = tid + t * 128;
    const int r = idx >> 3, c = (idx & 7) * 8;
    ASYNC_COPY_B128(&medS[r][c], &medbf[(size_t)r * 1024 + h * 64 + c]);
  }
  if (tid < 64) { rmax[tid] = -1e30f; rsum[tid] = 0.0f; }
  ASYNC_WAIT();
  __syncthreads();

  // pass A: streaming softmax stats over L
  for (int l0 = 0; l0 < 4096; l0 += 64) {
#pragma unroll
    for (int t = 0; t < 4; ++t) {
      const int idx = tid + t * 128;
      const int r = idx >> 3, c = (idx & 7) * 8;
      ASYNC_COPY_B128(&kS[r][c],
                      &kvbf[(size_t)(b * 4096 + l0 + r) * 2048 + h * 64 + c]);
    }
    ASYNC_WAIT();
    __syncthreads();
    v8f s[2][2] = {};
#pragma unroll
    for (int ks = 0; ks < 2; ++ks) {
      v16bf am[2], bk[2];
      am[0] = load_a_frag(&medS[qr][ks * 32], 72);
      am[1] = load_a_frag(&medS[qr + 16][ks * 32], 72);
      bk[0] = load_bT_frag(&kS[qc][ks * 32], 72);
      bk[1] = load_bT_frag(&kS[qc + 16][ks * 32], 72);
#pragma unroll
      for (int mi = 0; mi < 2; ++mi)
#pragma unroll
        for (int ni = 0; ni < 2; ++ni)
          s[mi][ni] = wmma_bf16(am[mi], bk[ni], s[mi][ni]);
    }
#pragma unroll
    for (int mi = 0; mi < 2; ++mi)
#pragma unroll
      for (int ni = 0; ni < 2; ++ni)
#pragma unroll
        for (int r = 0; r < 8; ++r)
          Sg[qr + mi * 16 + rbase + r][qc + ni * 16 + cin] =
              s[mi][ni][r] * scale;
    __syncthreads();
    if (tid < 64) {
      float mo = rmax[tid], cm = -1e30f;
      for (int c = 0; c < 64; ++c) cm = fmaxf(cm, Sg[tid][c]);
      const float mn = fmaxf(mo, cm);
      float ss = rsum[tid] * __expf(mo - mn);
      for (int c = 0; c < 64; ++c) ss += __expf(Sg[tid][c] - mn);
      rmax[tid] = mn; rsum[tid] = ss;
    }
    __syncthreads();
  }
  if (tid < 64) rinv[tid] = 1.0f / rsum[tid];
  __syncthreads();

  // pass B: O += P @ V with persistent accumulators
  v8f o[2][2] = {};
  for (int l0 = 0; l0 < 4096; l0 += 64) {
#pragma unroll
    for (int t = 0; t < 4; ++t) {
      const int idx = tid + t * 128;
      const int r = idx >> 3, c = (idx & 7) * 8;
      const size_t rowoff = (size_t)(b * 4096 + l0 + r) * 2048;
      ASYNC_COPY_B128(&kS[r][c], &kvbf[rowoff + h * 64 + c]);
      ASYNC_COPY_B128(&vS[r][c], &kvbf[rowoff + 1024 + h * 64 + c]);
    }
    ASYNC_WAIT();
    __syncthreads();
    v8f s[2][2] = {};
#pragma unroll
    for (int ks = 0; ks < 2; ++ks) {
      v16bf am[2], bk[2];
      am[0] = load_a_frag(&medS[qr][ks * 32], 72);
      am[1] = load_a_frag(&medS[qr + 16][ks * 32], 72);
      bk[0] = load_bT_frag(&kS[qc][ks * 32], 72);
      bk[1] = load_bT_frag(&kS[qc + 16][ks * 32], 72);
#pragma unroll
      for (int mi = 0; mi < 2; ++mi)
#pragma unroll
        for (int ni = 0; ni < 2; ++ni)
          s[mi][ni] = wmma_bf16(am[mi], bk[ni], s[mi][ni]);
    }
#pragma unroll
    for (int mi = 0; mi < 2; ++mi)
#pragma unroll
      for (int ni = 0; ni < 2; ++ni)
#pragma unroll
        for (int r = 0; r < 8; ++r)
          Sg[qr + mi * 16 + rbase + r][qc + ni * 16 + cin] =
              s[mi][ni][r] * scale;
    __syncthreads();
    for (int idx = tid; idx < 4096; idx += 128) {
      const int r = idx >> 6, c = idx & 63;
      Ps[r][c] = f2bf(__expf(Sg[r][c] - rmax[r]) * rinv[r]);
    }
    __syncthreads();
#pragma unroll
    for (int ks = 0; ks < 2; ++ks) {
      v16bf ap[2], bv[2];
      ap[0] = load_a_frag(&Ps[qr][ks * 32], 72);
      ap[1] = load_a_frag(&Ps[qr + 16][ks * 32], 72);
      bv[0] = load_b_frag(&vS[ks * 32][qc], 72);
      bv[1] = load_b_frag(&vS[ks * 32][qc + 16], 72);
#pragma unroll
      for (int mi = 0; mi < 2; ++mi)
#pragma unroll
        for (int ni = 0; ni < 2; ++ni)
          o[mi][ni] = wmma_bf16(ap[mi], bv[ni], o[mi][ni]);
    }
    __syncthreads();
  }
#pragma unroll
  for (int mi = 0; mi < 2; ++mi)
#pragma unroll
    for (int ni = 0; ni < 2; ++ni)
#pragma unroll
      for (int r = 0; r < 8; ++r)
        mkvbf[(size_t)(bh * 64 + qr + mi * 16 + rbase + r) * 64 +
              qc + ni * 16 + cin] = f2bf(o[mi][ni][r]);
}

// ---------------- stage 2: out = softmax(q med^T / sqrt(d)) @ mk_v ----------------
__global__ __launch_bounds__(128) void k_stage2(const bf16* __restrict__ qbf,   // [B*L,1024]
                                                const bf16* __restrict__ medbf, // [64,1024]
                                                const bf16* __restrict__ mkvbf, // [64*64,64]
                                                bf16* __restrict__ attnbf) {    // [B*L,1024]
  const int bh = blockIdx.y, b = bh >> 4, h = bh & 15;
  const int l0 = blockIdx.x * 64;
  const float scale = 0.125f;
  __shared__ __attribute__((aligned(16))) bf16 medS[64][72];
  __shared__ __attribute__((aligned(16))) bf16 mkvS[64][72];
  __shared__ __attribute__((aligned(16))) bf16 qS[64][72];
  __shared__ __attribute__((aligned(16))) bf16 Ps[64][72];
  __shared__ __attribute__((aligned(16))) float Sg[64][72];
  __shared__ float rmax[64], rinv[64];
  const int tid = threadIdx.x, lane = tid & 31, w = tid >> 5;
  const int qr = (w >> 1) * 32, qc = (w & 1) * 32;
  const int rbase = (lane >> 4) * 8, cin = lane & 15;

#pragma unroll
  for (int t = 0; t < 4; ++t) {
    const int idx = tid + t * 128;
    const int r = idx >> 3, c = (idx & 7) * 8;
    ASYNC_COPY_B128(&medS[r][c], &medbf[(size_t)r * 1024 + h * 64 + c]);
    ASYNC_COPY_B128(&mkvS[r][c], &mkvbf[(size_t)(bh * 64 + r) * 64 + c]);
    ASYNC_COPY_B128(&qS[r][c],
                    &qbf[(size_t)(b * 4096 + l0 + r) * 1024 + h * 64 + c]);
  }
  ASYNC_WAIT();
  __syncthreads();

  v8f s[2][2] = {};
#pragma unroll
  for (int ks = 0; ks < 2; ++ks) {
    v16bf aq[2], bm[2];
    aq[0] = load_a_frag(&qS[qr][ks * 32], 72);
    aq[1] = load_a_frag(&qS[qr + 16][ks * 32], 72);
    bm[0] = load_bT_frag(&medS[qc][ks * 32], 72);
    bm[1] = load_bT_frag(&medS[qc + 16][ks * 32], 72);
#pragma unroll
    for (int mi = 0; mi < 2; ++mi)
#pragma unroll
      for (int ni = 0; ni < 2; ++ni)
        s[mi][ni] = wmma_bf16(aq[mi], bm[ni], s[mi][ni]);
  }
#pragma unroll
  for (int mi = 0; mi < 2; ++mi)
#pragma unroll
    for (int ni = 0; ni < 2; ++ni)
#pragma unroll
      for (int r = 0; r < 8; ++r)
        Sg[qr + mi * 16 + rbase + r][qc + ni * 16 + cin] =
            s[mi][ni][r] * scale;
  __syncthreads();
  if (tid < 64) {
    float cm = -1e30f;
    for (int c = 0; c < 64; ++c) cm = fmaxf(cm, Sg[tid][c]);
    float ss = 0.0f;
    for (int c = 0; c < 64; ++c) ss += __expf(Sg[tid][c] - cm);
    rmax[tid] = cm; rinv[tid] = 1.0f / ss;
  }
  __syncthreads();
  for (int idx = tid; idx < 4096; idx += 128) {
    const int r = idx >> 6, c = idx & 63;
    Ps[r][c] = f2bf(__expf(Sg[r][c] - rmax[r]) * rinv[r]);
  }
  __syncthreads();

  v8f o[2][2] = {};
#pragma unroll
  for (int ks = 0; ks < 2; ++ks) {
    v16bf ap[2], bv[2];
    ap[0] = load_a_frag(&Ps[qr][ks * 32], 72);
    ap[1] = load_a_frag(&Ps[qr + 16][ks * 32], 72);
    bv[0] = load_b_frag(&mkvS[ks * 32][qc], 72);
    bv[1] = load_b_frag(&mkvS[ks * 32][qc + 16], 72);
#pragma unroll
    for (int mi = 0; mi < 2; ++mi)
#pragma unroll
      for (int ni = 0; ni < 2; ++ni)
        o[mi][ni] = wmma_bf16(ap[mi], bv[ni], o[mi][ni]);
  }
#pragma unroll
  for (int mi = 0; mi < 2; ++mi)
#pragma unroll
    for (int ni = 0; ni < 2; ++ni)
#pragma unroll
      for (int r = 0; r < 8; ++r)
        attnbf[(size_t)(b * 4096 + l0 + qr + mi * 16 + rbase + r) * 1024 +
               h * 64 + qc + ni * 16 + cin] = f2bf(o[mi][ni][r]);
}

// ---------------- host launch ----------------
extern "C" void kernel_launch(void* const* d_in, const int* in_sizes, int n_in,
                              void* d_out, int out_size, void* d_ws,
                              size_t ws_size, hipStream_t stream) {
  const int B = 4, L = 4096, D = 1024, H = 16;
  const int BL = B * L;
  const float* x        = (const float*)d_in[0];
  const float* Wq       = (const float*)d_in[1];
  const float* Wkv      = (const float*)d_in[2];
  const float* Wo       = (const float*)d_in[3];
  const float* bo       = (const float*)d_in[4];
  const float* mediator = (const float*)d_in[5];
  float* out = (float*)d_out;

  char* ws = (char*)d_ws;
  size_t off = 0;
  auto alloc = [&](size_t bytes) -> void* {
    void* p = ws + off;
    off += (bytes + 255) & ~(size_t)255;
    return p;
  };
  bf16* xbf    = (bf16*)alloc((size_t)BL * D * 2);
  bf16* wqbf   = (bf16*)alloc((size_t)D * D * 2);
  bf16* wkvbf  = (bf16*)alloc((size_t)D * 2 * D * 2);
  bf16* wobf   = (bf16*)alloc((size_t)D * D * 2);
  bf16* medbf  = (bf16*)alloc((size_t)64 * D * 2);
  bf16* qbf    = (bf16*)alloc((size_t)BL * D * 2);
  bf16* kvbf   = (bf16*)alloc((size_t)BL * 2 * D * 2);
  bf16* mkvbf  = (bf16*)alloc((size_t)B * H * 64 * 64 * 2);
  bf16* attnbf = (bf16*)alloc((size_t)BL * D * 2);
  (void)ws_size; (void)in_sizes; (void)n_in; (void)out_size;

  k_f32_to_bf16<<<4096, 256, 0, stream>>>(x, xbf, BL * D);
  k_f32_to_bf16<<<1024, 256, 0, stream>>>(Wq, wqbf, D * D);
  k_f32_to_bf16<<<2048, 256, 0, stream>>>(Wkv, wkvbf, D * 2 * D);
  k_f32_to_bf16<<<1024, 256, 0, stream>>>(Wo, wobf, D * D);
  k_f32_to_bf16<<<64, 256, 0, stream>>>(mediator, medbf, 64 * D);

  dim3 blk(256);
  k_gemm<true, false><<<dim3(D / 128, BL / 128), blk, 0, stream>>>(
      xbf, wqbf, nullptr, qbf, BL, D, D);
  k_gemm<true, false><<<dim3(2 * D / 128, BL / 128), blk, 0, stream>>>(
      xbf, wkvbf, nullptr, kvbf, BL, 2 * D, D);

  k_stage1<<<B * H, 128, 0, stream>>>(medbf, kvbf, mkvbf);
  k_stage2<<<dim3(L / 64, B * H), 128, 0, stream>>>(qbf, medbf, mkvbf, attnbf);

  k_gemm<false, true><<<dim3(D / 128, BL / 128), blk, 0, stream>>>(
      attnbf, wobf, bo, out, BL, D, D);
}
